// GSTVQA_4870492913833
// MI455X (gfx1250) — matrix operand order repaired and checked by reading, not compiled, not run on hardware
//
#include <hip/hip_runtime.h>
#include <hip/hip_bf16.h>

typedef unsigned short ushort_t;
typedef unsigned int uint_t;

typedef __attribute__((ext_vector_type(16))) __bf16 v16bf;
typedef __attribute__((ext_vector_type(8)))  float  v8f;

#define BB   64
#define TT   300
#define FF2  2944
#define FF1  1472
#define REDU 256
#define HH   32
#define KK   15
#define CAR  320
#define G3   96
#define MT   (BB * TT)        // 19200 rows
#define NCROP (TT - 2*(KK-1)) // 272
#define NA    (TT - (KK-1))   // 286

// ---------- helpers ----------
__device__ __forceinline__ ushort_t f2bf(float f) {
    uint_t u = __float_as_uint(f);
    uint_t r = u + 0x7FFFu + ((u >> 16) & 1u);   // round-to-nearest-even
    return (ushort_t)(r >> 16);
}
__device__ __forceinline__ float bf2f(ushort_t h) {
    return __uint_as_float(((uint_t)h) << 16);
}
__device__ __forceinline__ float sigmf(float x) {
    return 1.0f / (1.0f + __expf(-x));
}

union V16U { uint4 u[2]; v16bf v; };

// ---------- kernel 0: fp32 -> bf16 convert ----------
__global__ void k_f32_to_bf16(const float* __restrict__ in, ushort_t* __restrict__ out, long long n) {
    long long i = (long long)blockIdx.x * blockDim.x + threadIdx.x;
    long long s = (long long)gridDim.x * blockDim.x;
    for (; i < n; i += s) out[i] = f2bf(in[i]);
}

// ---------- kernel 1: normalize input -> bf16 xn ----------
__global__ void k_normalize(const float* __restrict__ x,
                            const float* __restrict__ mv, const float* __restrict__ sv,
                            const float* __restrict__ mm, const float* __restrict__ sm,
                            ushort_t* __restrict__ xn, long long n) {
    long long i = (long long)blockIdx.x * blockDim.x + threadIdx.x;
    long long s = (long long)gridDim.x * blockDim.x;
    for (; i < n; i += s) {
        int c = (int)(i % FF2);
        float v;
        if (c < FF1) v = (x[i] - mv[c]) / sv[c];
        else         v = (x[i] - mm[c - FF1]) / sm[c - FF1];
        xn[i] = f2bf(v);
    }
}

// ---------- kernel 2: WMMA bf16 GEMM  C[M,N] = A[M,K] * W[N,K]^T + bias ----------
// Software-pipelined over K (2 fragment sets, K % 64 == 0 for all call sites).
// epi: 0 = relu->bf16, 1 = sigmoid->bf16, 2 = bias->bf16, 3 = bias->f32
template <int NSUB>
__global__ __launch_bounds__(256) void k_gemm_wmma(const ushort_t* __restrict__ A,
                                                   const ushort_t* __restrict__ W,
                                                   const float* __restrict__ bias,
                                                   void* __restrict__ Cout,
                                                   int M, int N, int K, int epi) {
    int wave = (blockIdx.x * 256 + threadIdx.x) >> 5;
    int lane = threadIdx.x & 31;
    int mTiles = M >> 4;
    int strips = N / (16 * NSUB);
    if (wave >= mTiles * strips) return;
    int m0 = (wave % mTiles) << 4;
    int n0 = (wave / mTiles) * (16 * NSUB);

    int lr = lane & 15;
    int hi = lane >> 4;   // 0: lanes 0-15, 1: lanes 16-31

    v8f acc[NSUB];
#pragma unroll
    for (int s = 0; s < NSUB; ++s)
#pragma unroll
        for (int i = 0; i < 8; ++i) acc[s][i] = 0.0f;

    // A fragment: row m0+lr; lane-half picks K sub-offset {0,8}; chunks at +0 and +16
    const ushort_t* arow  = A + (size_t)(m0 + lr) * K + (hi ? 8 : 0);
    // B fragment: col n0+s*16+lr; lane-half picks K sub-offset {0,8}; chunks at +0 and +16
    const ushort_t* wbase = W + (size_t)(n0 + lr) * K + (hi ? 8 : 0);

    auto loadA = [&](V16U& a, int k0) {
        a.u[0] = *(const uint4*)(arow + k0);        // K = k0+{0..7}   (hi: +8..15)
        a.u[1] = *(const uint4*)(arow + k0 + 16);   // K = k0+{16..23} (hi: +24..31)
    };
    auto loadB = [&](V16U* b, int k0) {
#pragma unroll
        for (int s = 0; s < NSUB; ++s) {
            const ushort_t* wr = wbase + (size_t)s * 16 * K + k0;
            b[s].u[0] = *(const uint4*)(wr);        // K = k0+{0..7}   (hi: +8..15)
            b[s].u[1] = *(const uint4*)(wr + 16);   // K = k0+{16..23} (hi: +24..31)
        }
    };

    V16U a0, a1;
    V16U b0[NSUB], b1[NSUB];
    loadA(a0, 0);
    loadB(b0, 0);
    for (int k0 = 0; k0 < K; k0 += 64) {
        // stage 1 loads in flight while stage 0 computes
        loadA(a1, k0 + 32);
        loadB(b1, k0 + 32);
        __builtin_prefetch(arow + k0 + 512, 0, 1);   // speculative stream-ahead on A
#pragma unroll
        for (int s = 0; s < NSUB; ++s)
            acc[s] = __builtin_amdgcn_wmma_f32_16x16x32_bf16(
                false, a0.v, false, b0[s].v, (short)0, acc[s], false, false);
        int k2 = (k0 + 64 < K) ? (k0 + 64) : 0;      // wrap: harmless dummy reload on last iter
        loadA(a0, k2);
        loadB(b0, k2);
#pragma unroll
        for (int s = 0; s < NSUB; ++s)
            acc[s] = __builtin_amdgcn_wmma_f32_16x16x32_bf16(
                false, a1.v, false, b1[s].v, (short)0, acc[s], false, false);
    }

#pragma unroll
    for (int s = 0; s < NSUB; ++s) {
        int col = n0 + s * 16 + lr;
        float bv = bias[col];
#pragma unroll
        for (int i = 0; i < 8; ++i) {
            int row = m0 + i + (hi ? 8 : 0);
            float v = acc[s][i] + bv;
            if (epi == 0) v = fmaxf(v, 0.0f);
            else if (epi == 1) v = sigmf(v);
            size_t o = (size_t)row * N + col;
            if (epi == 3) ((float*)Cout)[o] = v;
            else          ((ushort_t*)Cout)[o] = f2bf(v);
        }
    }
}

// ---------- kernel 3: fuse channel-attention gate ----------
__global__ void k_fuse(const ushort_t* __restrict__ wsig, const ushort_t* __restrict__ xn,
                       ushort_t* __restrict__ feat, long long n) {
    long long i = (long long)blockIdx.x * blockDim.x + threadIdx.x;
    long long s = (long long)gridDim.x * blockDim.x;
    for (; i < n; i += s) {
        long long m = i / FF1;
        int c = (int)(i % FF1);
        size_t base = (size_t)m * FF2;
        float v = bf2f(wsig[base + c])       * bf2f(xn[base + c]) +
                  bf2f(wsig[base + FF1 + c]) * bf2f(xn[base + FF1 + c]);
        feat[i] = f2bf(v);
    }
}

// ---------- kernel 4: GRU scan (one wave32 per batch, one lane per hidden unit) ----------
__global__ __launch_bounds__(32) void k_gru(const float* __restrict__ w_hh,  // [96,32]
                                            const float* __restrict__ b_hh,  // [96]
                                            const float* __restrict__ xg,    // [MT,96] incl b_ih
                                            float* __restrict__ outs) {      // [MT,32]
    __shared__ float wsm[32 * 96];   // transposed: wsm[k*96 + row], conflict-free
    int b = blockIdx.x;
    int j = threadIdx.x;             // hidden unit 0..31
    for (int r = j; r < 96 * 32; r += 32) {
        int row = r / 32, k = r % 32;
        wsm[k * 96 + row] = w_hh[row * 32 + k];
    }
    __syncthreads();
    float bhr = b_hh[j], bhz = b_hh[32 + j], bhn = b_hh[64 + j];
    float h = 0.0f;
    for (int t = 0; t < TT; ++t) {
        const float* xgt = xg + ((size_t)b * TT + t) * G3;
        float xr = xgt[j], xz = xgt[32 + j], xnv = xgt[64 + j];
        float hr = bhr, hz = bhz, hn = bhn;
#pragma unroll
        for (int k = 0; k < 32; ++k) {
            float hk = __shfl(h, k, 32);
            hr += wsm[k * 96 + j]      * hk;
            hz += wsm[k * 96 + 32 + j] * hk;
            hn += wsm[k * 96 + 64 + j] * hk;
        }
        float r = sigmf(xr + hr);
        float z = sigmf(xz + hz);
        float n = tanhf(xnv + r * hn);
        h = (1.0f - z) * n + z * h;
        outs[((size_t)b * TT + t) * 32 + j] = h;
    }
}

// ---------- kernel 5: attention convs + pyramid pooling + heads ----------
__global__ __launch_bounds__(256) void k_head(const float* __restrict__ outs,
                                              const float* __restrict__ aw1, const float* __restrict__ ab1,
                                              const float* __restrict__ aw2, const float* __restrict__ ab2,
                                              const float* __restrict__ qw,  const float* __restrict__ qb,
                                              const float* __restrict__ rw,  const float* __restrict__ rb,
                                              float* __restrict__ score) {
    __shared__ float so[TT * HH];     // 37.5 KB of 320 KB WGP LDS
    __shared__ float a1[NA];
    __shared__ float att[NCROP];
    __shared__ float proj[NCROP];
    __shared__ float qe[127];
    int b = blockIdx.x, tid = threadIdx.x;

    for (int i = tid; i < TT * HH; i += 256) so[i] = outs[(size_t)b * TT * HH + i];
    __syncthreads();

    // conv1 H->1 (k=15) + ReLU
    for (int t = tid; t < NA; t += 256) {
        float s = ab1[0];
        for (int h = 0; h < HH; ++h) {
            const float* wv = aw1 + h * KK;
#pragma unroll
            for (int k = 0; k < KK; ++k) s += so[(t + k) * HH + h] * wv[k];
        }
        a1[t] = fmaxf(s, 0.0f);
    }
    __syncthreads();

    // conv2 1->1 (k=15) + tanh; projection of cropped features onto q_w
    for (int t = tid; t < NCROP; t += 256) {
        float s = ab2[0];
#pragma unroll
        for (int k = 0; k < KK; ++k) s += a1[t + k] * aw2[k];
        att[t] = tanhf(s);
        float p = 0.0f;
#pragma unroll
        for (int h = 0; h < HH; ++h) p += so[(t + KK - 1) * HH + h] * qw[h];
        proj[t] = p;   // q_fet[t] . q_w == att[t]*proj[t]
    }
    __syncthreads();

    // 127 rows: row0 = plain mean; rows 1..126 = pyramid bins (1/(e-s) cancels in num/den)
    for (int row = tid; row < 127; row += 256) {
        float qv;
        if (row == 0) {
            float s = 0.0f;
            for (int n = 0; n < NCROP; ++n) s += att[n] * proj[n];
            qv = s / (float)NCROP + qb[0];
        } else {
            int idx = row - 1, m = 2, base = 0;
            while (idx >= base + m) { base += m; m <<= 1; }
            int i = idx - base;
            int s0 = (i * NCROP) / m;
            int e0 = ((i + 1) * NCROP + m - 1) / m;
            float num = 0.0f, den = 0.0f;
            for (int n = s0; n < e0; ++n) { num += att[n] * proj[n]; den += att[n]; }
            qv = num / den + qb[0];
        }
        qe[row] = qv;
    }
    __syncthreads();

    if (tid == 0) {
        float s = rb[0];
        for (int r = 0; r < 127; ++r) s += qe[r] * rw[r];
        score[b] = s;
    }
}

// ---------- launch ----------
extern "C" void kernel_launch(void* const* d_in, const int* in_sizes, int n_in,
                              void* d_out, int out_size, void* d_ws, size_t ws_size,
                              hipStream_t stream) {
    const float* input    = (const float*)d_in[0];
    // d_in[1] = input_length (int64), unused (reference ignores it for full-length seqs)
    const float* mean_var = (const float*)d_in[2];
    const float* std_var  = (const float*)d_in[3];
    const float* mean_mean= (const float*)d_in[4];
    const float* std_mean = (const float*)d_in[5];
    const float* ca_fc1_w = (const float*)d_in[6];
    const float* ca_fc1_b = (const float*)d_in[7];
    const float* ca_fc2_w = (const float*)d_in[8];
    const float* ca_fc2_b = (const float*)d_in[9];
    const float* ann_w    = (const float*)d_in[10];
    const float* ann_b    = (const float*)d_in[11];
    const float* gru_w_ih = (const float*)d_in[12];
    const float* gru_w_hh = (const float*)d_in[13];
    const float* gru_b_ih = (const float*)d_in[14];
    const float* gru_b_hh = (const float*)d_in[15];
    const float* q_w      = (const float*)d_in[16];
    const float* q_b      = (const float*)d_in[17];
    const float* att_w1   = (const float*)d_in[18];
    const float* att_b1   = (const float*)d_in[19];
    const float* att_w2   = (const float*)d_in[20];
    const float* att_b2   = (const float*)d_in[21];
    const float* qreg2_w  = (const float*)d_in[22];
    const float* qreg2_b  = (const float*)d_in[23];

    char* ws = (char*)d_ws;
    size_t off = 0;
    auto take = [&](size_t bytes) { char* p = ws + off; off += (bytes + 255) & ~(size_t)255; return p; };

    ushort_t* xn_bf   = (ushort_t*)take((size_t)2 * MT * FF2);
    ushort_t* wfc1_bf = (ushort_t*)take((size_t)2 * CAR * FF2);
    ushort_t* h1_bf   = (ushort_t*)take((size_t)2 * MT * CAR);
    ushort_t* wfc2_bf = (ushort_t*)take((size_t)2 * FF2 * CAR);
    ushort_t* wsig_bf = (ushort_t*)take((size_t)2 * MT * FF2);
    ushort_t* feat_bf = (ushort_t*)take((size_t)2 * MT * FF1);
    ushort_t* wann_bf = (ushort_t*)take((size_t)2 * REDU * FF1);
    ushort_t* ann_bf  = (ushort_t*)take((size_t)2 * MT * REDU);
    ushort_t* wih_bf  = (ushort_t*)take((size_t)2 * G3 * REDU);
    float*    xg_f    = (float*)take((size_t)4 * MT * G3);
    float*    outs_f  = (float*)take((size_t)4 * MT * HH);

    // weight conversions (tiny)
    k_f32_to_bf16<<<256, 256, 0, stream>>>(ca_fc1_w, wfc1_bf, (long long)CAR * FF2);
    k_f32_to_bf16<<<256, 256, 0, stream>>>(ca_fc2_w, wfc2_bf, (long long)FF2 * CAR);
    k_f32_to_bf16<<<128, 256, 0, stream>>>(ann_w,    wann_bf, (long long)REDU * FF1);
    k_f32_to_bf16<<<32,  256, 0, stream>>>(gru_w_ih, wih_bf,  (long long)G3 * REDU);

    // normalize -> bf16
    k_normalize<<<4096, 256, 0, stream>>>(input, mean_var, std_var, mean_mean, std_mean,
                                          xn_bf, (long long)MT * FF2);

    // GEMM1: xn[19200,2944] * fc1[320,2944]^T -> relu -> h1 bf16
    {
        int waves = (MT / 16) * (CAR / 64);
        k_gemm_wmma<4><<<(waves + 7) / 8, 256, 0, stream>>>(xn_bf, wfc1_bf, ca_fc1_b,
                                                            h1_bf, MT, CAR, FF2, 0);
    }
    // GEMM2: h1[19200,320] * fc2[2944,320]^T -> sigmoid -> wsig bf16
    {
        int waves = (MT / 16) * (FF2 / 64);
        k_gemm_wmma<4><<<(waves + 7) / 8, 256, 0, stream>>>(h1_bf, wfc2_bf, ca_fc2_b,
                                                            wsig_bf, MT, FF2, CAR, 1);
    }
    // fuse gate: feat = w_var*x_var + w_mean*x_mean
    k_fuse<<<4096, 256, 0, stream>>>(wsig_bf, xn_bf, feat_bf, (long long)MT * FF1);

    // GEMM3: feat[19200,1472] * ann[256,1472]^T + b -> ann bf16
    {
        int waves = (MT / 16) * (REDU / 64);
        k_gemm_wmma<4><<<(waves + 7) / 8, 256, 0, stream>>>(feat_bf, wann_bf, ann_b,
                                                            ann_bf, MT, REDU, FF1, 2);
    }
    // GEMM4: ann[19200,256] * w_ih[96,256]^T + b_ih -> xg f32
    {
        int waves = (MT / 16) * (G3 / 32);
        k_gemm_wmma<2><<<(waves + 7) / 8, 256, 0, stream>>>(ann_bf, wih_bf, gru_b_ih,
                                                            xg_f, MT, G3, REDU, 3);
    }

    // GRU scan
    k_gru<<<BB, 32, 0, stream>>>(gru_w_hh, gru_b_hh, xg_f, outs_f);

    // attention + pyramid + heads -> score [B,1]
    k_head<<<BB, 256, 0, stream>>>(outs_f, att_w1, att_b1, att_w2, att_b2,
                                   q_w, q_b, qreg2_w, qreg2_b, (float*)d_out);
}
